// LlamaAttention_23974507446451
// MI455X (gfx1250) — compile-verified, matrix operand order checked by reading, not run on hardware
//
#include <hip/hip_runtime.h>
#include <hip/hip_bf16.h>
#include <cstdint>

// ---------------------------------------------------------------------------
// MI455X (gfx1250) Llama attention block: bf16 WMMA everywhere, async-to-LDS
// double-buffered GEMMs, flash attention with online softmax.
// ---------------------------------------------------------------------------

#define T_SEQ   2048
#define HID     4096
#define NHEADS  32
#define NKV     8
#define HDIM    128
#define QKV_N   6144   // (32 + 2*8) * 128

typedef __attribute__((ext_vector_type(16))) __bf16 v16bf;
typedef __attribute__((ext_vector_type(8)))  float  v8f;

union Frag {
    v16bf    bf;
    uint32_t u[8];
};

__device__ __forceinline__ uint16_t f2bf(float f) {
    uint32_t u = __builtin_bit_cast(uint32_t, f);
    u += 0x7FFFu + ((u >> 16) & 1u);          // round-to-nearest-even
    return (uint16_t)(u >> 16);
}
__device__ __forceinline__ uint32_t pack2(uint16_t lo, uint16_t hi) {
    return (uint32_t)lo | ((uint32_t)hi << 16);
}

__device__ __forceinline__ v8f wmma_bf16(const Frag& a, const Frag& b, v8f c) {
    return __builtin_amdgcn_wmma_f32_16x16x32_bf16(
        false, a.bf, false, b.bf, (short)0, c, false, false);
}

// CDNA5 async HBM->LDS copy, 16B per lane, tracked by ASYNCcnt.
// LDS aperture keeps the byte offset in the flat address's low 32 bits.
__device__ __forceinline__ void async_b128(const void* lds_dst, const void* gsrc) {
    uint32_t ldsoff = (uint32_t)(uintptr_t)lds_dst;
    asm volatile("global_load_async_to_lds_b128 %0, %1, off"
                 :: "v"(ldsoff), "v"(gsrc) : "memory");
}
__device__ __forceinline__ void wait_async0() {
    asm volatile("s_wait_asynccnt 0x0" ::: "memory");
}

// ---------------------------------------------------------------------------
// fp32 -> bf16 elementwise (8 elems / thread, 16B stores)
// ---------------------------------------------------------------------------
__global__ __launch_bounds__(256) void convert_bf16_kernel(
    const float4* __restrict__ src, uint4* __restrict__ dst, int n8)
{
    const int i = blockIdx.x * 256 + threadIdx.x;
    if (i >= n8) return;
    const float4 a = src[2 * i], b = src[2 * i + 1];
    uint4 o;
    o.x = pack2(f2bf(a.x), f2bf(a.y));
    o.y = pack2(f2bf(a.z), f2bf(a.w));
    o.z = pack2(f2bf(b.x), f2bf(b.y));
    o.w = pack2(f2bf(b.z), f2bf(b.w));
    dst[i] = o;
}

// ---------------------------------------------------------------------------
// fp32 W[K][N] -> bf16 WT[N][K] via 32x32 LDS tile transpose
// ---------------------------------------------------------------------------
__global__ __launch_bounds__(256) void transpose_bf16_kernel(
    const float* __restrict__ W, uint16_t* __restrict__ WT, int K, int N)
{
    __shared__ uint16_t Ts[32][33];
    const int n0 = blockIdx.x * 32, k0 = blockIdx.y * 32;
    const int tid = threadIdx.x;
    #pragma unroll
    for (int j = 0; j < 4; ++j) {
        const int i = tid + j * 256;
        const int r = i >> 5, c = i & 31;          // r: k, c: n (coalesced in n)
        Ts[r][c] = f2bf(W[(size_t)(k0 + r) * N + n0 + c]);
    }
    __syncthreads();
    #pragma unroll
    for (int j = 0; j < 4; ++j) {
        const int i = tid + j * 256;
        const int rn = i >> 5, ck = i & 31;        // coalesced in k
        WT[(size_t)(n0 + rn) * K + k0 + ck] = Ts[ck][rn];
    }
}

// ---------------------------------------------------------------------------
// GEMM: C[M,N](f32) = A[M,K](bf16) * BT[N,K](bf16, pre-transposed).
// 128x128 block tile, 8 waves as 4(M)x2(N), wave tile 32x64 = 8 WMMA / K-step.
// Double-buffered LDS filled with global_load_async_to_lds_b128 (ASYNCcnt).
// ---------------------------------------------------------------------------
__global__ __launch_bounds__(256) void gemm_async_kernel(
    const uint16_t* __restrict__ A, const uint16_t* __restrict__ BT,
    float* __restrict__ C, int M, int N, int K)
{
    constexpr int TM = 128, TN = 128, TK = 32, LDSK = 40; // 80B row: 16B aligned
    __shared__ uint16_t As[2][TM][LDSK];
    __shared__ uint16_t Bs[2][TN][LDSK];

    const int tid  = threadIdx.x;
    const int lane = tid & 31;
    const int wave = tid >> 5;
    const int hf   = lane >> 4;
    const int l16  = lane & 15;
    const int mbase = blockIdx.y * TM;
    const int nbase = blockIdx.x * TN;
    const int wm = (wave >> 1) * 32;   // 4 wave-rows of 32
    const int wn = (wave & 1) * 64;    // 2 wave-cols of 64

    // per-thread async copy slots: 16B (8 halves) each; rows crow and crow+64
    const int crow = tid >> 2;
    const int ckc  = (tid & 3) * 8;

    v8f acc[2][4];
    #pragma unroll
    for (int s = 0; s < 2; ++s)
        #pragma unroll
        for (int t = 0; t < 4; ++t)
            #pragma unroll
            for (int r = 0; r < 8; ++r) acc[s][t][r] = 0.f;

    auto issue = [&](int buf, int kb) {
        const uint16_t* ga = A + (size_t)(mbase + crow) * K + kb + ckc;
        async_b128(&As[buf][crow][ckc],      ga);
        async_b128(&As[buf][crow + 64][ckc], ga + (size_t)64 * K);
        const uint16_t* gb = BT + (size_t)(nbase + crow) * K + kb + ckc;
        async_b128(&Bs[buf][crow][ckc],      gb);
        async_b128(&Bs[buf][crow + 64][ckc], gb + (size_t)64 * K);
    };

    issue(0, 0);
    int p = 0;
    for (int kb = 0; kb < K; kb += TK) {
        wait_async0();          // this wave's copies into buf p have landed
        __syncthreads();        // everyone's copies have landed
        if (kb + TK < K) issue(p ^ 1, kb + TK);   // overlap fill with compute

        Frag a[2];
        #pragma unroll
        for (int s = 0; s < 2; ++s) {
            const int arow = wm + s * 16 + l16;
            #pragma unroll
            for (int i = 0; i < 4; ++i) {
                a[s].u[i]     = *(const uint32_t*)&As[p][arow][hf * 8 + 2 * i];
                a[s].u[4 + i] = *(const uint32_t*)&As[p][arow][16 + hf * 8 + 2 * i];
            }
        }
        #pragma unroll
        for (int t = 0; t < 4; ++t) {
            Frag b;
            const int ncol = wn + t * 16 + l16;
            #pragma unroll
            for (int i = 0; i < 4; ++i) {
                b.u[i]     = *(const uint32_t*)&Bs[p][ncol][hf * 8 + 2 * i];
                b.u[4 + i] = *(const uint32_t*)&Bs[p][ncol][16 + hf * 8 + 2 * i];
            }
            acc[0][t] = wmma_bf16(a[0], b, acc[0][t]);
            acc[1][t] = wmma_bf16(a[1], b, acc[1][t]);
        }
        __syncthreads();        // all reads of buf p done before it is refilled
        p ^= 1;
    }

    #pragma unroll
    for (int s = 0; s < 2; ++s)
        #pragma unroll
        for (int t = 0; t < 4; ++t)
            #pragma unroll
            for (int r = 0; r < 8; ++r)
                C[(size_t)(mbase + wm + s * 16 + r + 8 * hf) * N +
                  nbase + wn + t * 16 + l16] = acc[s][t][r];
}

// ---------------------------------------------------------------------------
// RoPE + layout conversion:
//   q -> bf16 [h][t][d] scaled by 1/sqrt(128);  k -> bf16 [g][t][d]
//   v -> bf16 transposed [g][d][t]
// ---------------------------------------------------------------------------
__global__ __launch_bounds__(128) void rope_kernel(
    const float* __restrict__ qkv, const float* __restrict__ cosp,
    const float* __restrict__ sinp, uint16_t* __restrict__ qb,
    uint16_t* __restrict__ kb, uint16_t* __restrict__ vT)
{
    const int t = blockIdx.x;
    const int h = blockIdx.y;     // 0..31 q, 32..39 k, 40..47 v
    const int d = threadIdx.x;
    const float* row = qkv + (size_t)t * QKV_N;

    if (h < NHEADS) {
        const float x  = row[h * HDIM + d];
        const float xr = (d < 64) ? -row[h * HDIM + d + 64] : row[h * HDIM + d - 64];
        const float c  = cosp[t * HDIM + d], s = sinp[t * HDIM + d];
        const float vq = (x * c + xr * s) * 0.08838834764831845f; // 1/sqrt(128)
        qb[((size_t)h * T_SEQ + t) * HDIM + d] = f2bf(vq);
    } else if (h < NHEADS + NKV) {
        const int g = h - NHEADS;
        const float* kr = row + NHEADS * HDIM + g * HDIM;
        const float x  = kr[d];
        const float xr = (d < 64) ? -kr[d + 64] : kr[d - 64];
        const float c  = cosp[t * HDIM + d], s = sinp[t * HDIM + d];
        kb[((size_t)g * T_SEQ + t) * HDIM + d] = f2bf(x * c + xr * s);
    } else {
        const int g = h - NHEADS - NKV;
        const float x = row[(NHEADS + NKV) * HDIM + g * HDIM + d];
        vT[((size_t)g * HDIM + d) * T_SEQ + t] = f2bf(x);
    }
}

// ---------------------------------------------------------------------------
// Flash attention, causal, GQA (R=4). One wave per 16-query tile, online
// softmax over 32-key chunks, all matmuls via v_wmma_f32_16x16x32_bf16.
// ---------------------------------------------------------------------------
__global__ __launch_bounds__(128) void attn_kernel(
    const uint16_t* __restrict__ qb, const uint16_t* __restrict__ kb,
    const uint16_t* __restrict__ vT, uint16_t* __restrict__ ao)
{
    __shared__ uint16_t Plds[4][16][36];   // per-wave 16x32 bf16 P

    const int tid  = threadIdx.x;
    const int lane = tid & 31;
    const int wave = tid >> 5;
    const int hf   = lane >> 4;
    const int l16  = lane & 15;
    const int h    = blockIdx.y;
    const int g    = h >> 2;                       // kv head (R = 4)
    const int qbase = blockIdx.x * 64 + wave * 16;

    Frag qf[4];
    {
        const uint16_t* qrow = qb + ((size_t)h * T_SEQ + qbase + l16) * HDIM;
        #pragma unroll
        for (int c = 0; c < 4; ++c)
            #pragma unroll
            for (int i = 0; i < 4; ++i) {
                qf[c].u[i]     = *(const uint32_t*)(qrow + 32 * c + hf * 8 + 2 * i);
                qf[c].u[4 + i] = *(const uint32_t*)(qrow + 32 * c + 16 + hf * 8 + 2 * i);
            }
    }

    float m[8], l[8];
    v8f acc[8];
    #pragma unroll
    for (int r = 0; r < 8; ++r) { m[r] = -1e30f; l[r] = 0.f; }
    #pragma unroll
    for (int t = 0; t < 8; ++t)
        #pragma unroll
        for (int r = 0; r < 8; ++r) acc[t][r] = 0.f;

    const uint16_t* kh = kb + (size_t)g * T_SEQ * HDIM;
    const uint16_t* vh = vT + (size_t)g * HDIM * T_SEQ;

    const int nchunks = (qbase + 47) >> 5;   // keys 0 .. qbase+15
    for (int cb = 0; cb < nchunks; ++cb) {
        const int sbase = cb * 32;

        v8f S0, S1;
        #pragma unroll
        for (int r = 0; r < 8; ++r) { S0[r] = 0.f; S1[r] = 0.f; }
        {
            const uint16_t* kr0 = kh + (size_t)(sbase + l16) * HDIM;
            const uint16_t* kr1 = kr0 + 16 * HDIM;
            #pragma unroll
            for (int c = 0; c < 4; ++c) {
                Frag b0, b1;
                #pragma unroll
                for (int i = 0; i < 4; ++i) {
                    b0.u[i]     = *(const uint32_t*)(kr0 + 32 * c + hf * 8 + 2 * i);
                    b0.u[4 + i] = *(const uint32_t*)(kr0 + 32 * c + 16 + hf * 8 + 2 * i);
                    b1.u[i]     = *(const uint32_t*)(kr1 + 32 * c + hf * 8 + 2 * i);
                    b1.u[4 + i] = *(const uint32_t*)(kr1 + 32 * c + 16 + hf * 8 + 2 * i);
                }
                S0 = wmma_bf16(qf[c], b0, S0);
                S1 = wmma_bf16(qf[c], b1, S1);
            }
        }

        if (sbase + 31 > qbase) {            // diagonal-crossing chunks only
            #pragma unroll
            for (int r = 0; r < 8; ++r) {
                const int tg = qbase + r + 8 * hf;
                S0[r] = (sbase + l16 > tg)      ? -1e30f : S0[r];
                S1[r] = (sbase + 16 + l16 > tg) ? -1e30f : S1[r];
            }
        }

        float mc[8];
        #pragma unroll
        for (int r = 0; r < 8; ++r) mc[r] = fmaxf(S0[r], S1[r]);
        #pragma unroll
        for (int off = 1; off < 16; off <<= 1)
            #pragma unroll
            for (int r = 0; r < 8; ++r)
                mc[r] = fmaxf(mc[r], __shfl_xor(mc[r], off, 32));

        float corr[8], P0[8], P1[8], rs[8];
        #pragma unroll
        for (int r = 0; r < 8; ++r) {
            const float mn = fmaxf(m[r], mc[r]);
            corr[r] = __expf(m[r] - mn);
            m[r]    = mn;
            P0[r]   = __expf(S0[r] - mn);
            P1[r]   = __expf(S1[r] - mn);
            rs[r]   = P0[r] + P1[r];
        }
        #pragma unroll
        for (int off = 1; off < 16; off <<= 1)
            #pragma unroll
            for (int r = 0; r < 8; ++r)
                rs[r] += __shfl_xor(rs[r], off, 32);
        #pragma unroll
        for (int r = 0; r < 8; ++r) l[r] = l[r] * corr[r] + rs[r];
        #pragma unroll
        for (int t = 0; t < 8; ++t)
            #pragma unroll
            for (int r = 0; r < 8; ++r) acc[t][r] *= corr[r];

        // transpose P (D-layout) -> A-layout through LDS (wave-local)
        #pragma unroll
        for (int r = 0; r < 8; ++r) {
            Plds[wave][8 * hf + r][l16]      = f2bf(P0[r]);
            Plds[wave][8 * hf + r][16 + l16] = f2bf(P1[r]);
        }
        asm volatile("s_wait_dscnt 0x0" ::: "memory");
        Frag pa;
        #pragma unroll
        for (int i = 0; i < 4; ++i) {
            pa.u[i]     = *(const uint32_t*)&Plds[wave][l16][hf * 8 + 2 * i];
            pa.u[4 + i] = *(const uint32_t*)&Plds[wave][l16][16 + hf * 8 + 2 * i];
        }
        asm volatile("" ::: "memory");

        #pragma unroll
        for (int t = 0; t < 8; ++t) {
            const uint16_t* vr = vh + (size_t)(t * 16 + l16) * T_SEQ + sbase;
            Frag vb;
            #pragma unroll
            for (int i = 0; i < 4; ++i) {
                vb.u[i]     = *(const uint32_t*)(vr + hf * 8 + 2 * i);
                vb.u[4 + i] = *(const uint32_t*)(vr + 16 + hf * 8 + 2 * i);
            }
            acc[t] = wmma_bf16(pa, vb, acc[t]);
        }
    }

    float inv[8];
    #pragma unroll
    for (int r = 0; r < 8; ++r) inv[r] = 1.f / l[r];
    #pragma unroll
    for (int t = 0; t < 8; ++t)
        #pragma unroll
        for (int r = 0; r < 8; ++r)
            ao[(size_t)(qbase + r + 8 * hf) * (NHEADS * HDIM) + h * HDIM + t * 16 + l16] =
                f2bf(acc[t][r] * inv[r]);
}

// ---------------------------------------------------------------------------
// Host launcher. Workspace regions are aliased where lifetimes are disjoint;
// every region is fully rewritten before being read on every call.
// ---------------------------------------------------------------------------
extern "C" void kernel_launch(void* const* d_in, const int* in_sizes, int n_in,
                              void* d_out, int out_size, void* d_ws, size_t ws_size,
                              hipStream_t stream) {
    (void)in_sizes; (void)n_in; (void)out_size; (void)ws_size;
    const float* hidden = (const float*)d_in[0];
    const float* cosp   = (const float*)d_in[1];
    const float* sinp   = (const float*)d_in[2];
    const float* w_qkv  = (const float*)d_in[3];
    const float* w_o    = (const float*)d_in[4];
    float* out = (float*)d_out;

    char* ws = (char*)d_ws;
    size_t off = 0;
    // region 0: hs bf16, later reused for q bf16 (same 16 MB)
    uint16_t* hsb   = (uint16_t*)(ws + off);
    uint16_t* qbuf  = (uint16_t*)(ws + off); off += (size_t)T_SEQ * HID * 2;
    // region 1: w_qkv^T bf16 (50 MB), later reused for attn_out bf16 (16 MB)
    uint16_t* wqkvT = (uint16_t*)(ws + off);
    uint16_t* ao    = (uint16_t*)(ws + off); off += (size_t)QKV_N * HID * 2;
    // region 2: qkv f32 (50 MB), later reused for w_o^T bf16 (32 MB)
    float*    qkv   = (float*)(ws + off);
    uint16_t* woT   = (uint16_t*)(ws + off); off += (size_t)T_SEQ * QKV_N * 4;
    uint16_t* kbuf  = (uint16_t*)(ws + off); off += (size_t)NKV * T_SEQ * HDIM * 2;
    uint16_t* vT    = (uint16_t*)(ws + off); off += (size_t)NKV * HDIM * T_SEQ * 2;

    // 1) hidden f32 -> bf16
    {
        const int n8 = T_SEQ * HID / 8;
        convert_bf16_kernel<<<n8 / 256, 256, 0, stream>>>(
            (const float4*)hidden, (uint4*)hsb, n8);
    }
    // 2) w_qkv -> bf16 transposed [N][K]
    transpose_bf16_kernel<<<dim3(QKV_N / 32, HID / 32), 256, 0, stream>>>(
        w_qkv, wqkvT, HID, QKV_N);
    // 3) QKV projection (async double-buffered bf16 WMMA)
    gemm_async_kernel<<<dim3(QKV_N / 128, T_SEQ / 128), 256, 0, stream>>>(
        hsb, wqkvT, qkv, T_SEQ, QKV_N, HID);
    // 4) RoPE + bf16 conversion (+V transpose, scale folded into Q)
    rope_kernel<<<dim3(T_SEQ, NHEADS + 2 * NKV), 128, 0, stream>>>(
        qkv, cosp, sinp, qbuf, kbuf, vT);
    // 5) w_o -> bf16 transposed (reuses qkv region, dead after RoPE)
    transpose_bf16_kernel<<<dim3(HID / 32, HID / 32), 256, 0, stream>>>(
        w_o, woT, HID, HID);
    // 6) causal flash attention
    attn_kernel<<<dim3(T_SEQ / 64, NHEADS), 128, 0, stream>>>(qbuf, kbuf, vT, ao);
    // 7) output projection
    gemm_async_kernel<<<dim3(HID / 128, T_SEQ / 128), 256, 0, stream>>>(
        ao, woT, out, T_SEQ, HID, HID);
}
